// DeepseekMLA_71683004170416
// MI455X (gfx1250) — matrix operand order repaired and checked by reading, not compiled, not run on hardware
//
#include <hip/hip_runtime.h>

// ---------------------------------------------------------------------------
// DeepSeek MLA prefill for MI455X (gfx1250): bf16-native, WMMA everywhere,
// register-pipelined global->LDS staging.
// ---------------------------------------------------------------------------

#define TT      2048
#define HIDDEN  7168
#define NH      32
#define DQK     192
#define DNOPE   128
#define DROPE   64
#define DV      128
#define QLORA   1536
#define KVLORA  512

typedef __attribute__((ext_vector_type(16))) __bf16 v16bf;
typedef __attribute__((ext_vector_type(8)))  float  v8f;

__device__ __forceinline__ unsigned short f2bf(float x) {
    union { float f; unsigned u; } v; v.f = x;
    unsigned u = v.u + 0x7FFFu + ((v.u >> 16) & 1u);   // round-to-nearest-even
    return (unsigned short)(u >> 16);
}
__device__ __forceinline__ float bf2f(unsigned short h) {
    union { unsigned u; float f; } v; v.u = ((unsigned)h) << 16; return v.f;
}

// Build a 16-element bf16 fragment from two 16-byte reads.
__device__ __forceinline__ v16bf ldfrag(const unsigned short* p0,
                                        const unsigned short* p1) {
    union { v16bf v; uint4 q[2]; } f;
    f.q[0] = *(const uint4*)p0;
    f.q[1] = *(const uint4*)p1;
    return f.v;
}

// ===========================================================================
// f32 -> bf16 bulk conversion (float4 in, 4xbf16 out)
// ===========================================================================
__global__ __launch_bounds__(256)
void mla_cvt_bf16(const float* __restrict__ x, unsigned short* __restrict__ y,
                  int n4)
{
    int i = blockIdx.x * 256 + threadIdx.x;
    if (i >= n4) return;
    float4 v = *(const float4*)(x + (size_t)i * 4);
    union { uint2 q; unsigned short h[4]; } u;
    u.h[0] = f2bf(v.x); u.h[1] = f2bf(v.y);
    u.h[2] = f2bf(v.z); u.h[3] = f2bf(v.w);
    *(uint2*)(y + (size_t)i * 4) = u.q;
}

// ===========================================================================
// GEMM: C[M,N] = A[M,K] @ B[K,N]; A,B bf16; C f32 or bf16.
// 128x128x64 tiles, 8 wave32 waves (4x2), wave tile 32x64 = 2x4 WMMA tiles,
// 2 k-chunks of 32 per stage (16 WMMAs / barrier pair).
// Staging is register-pipelined: next tile's global loads issue right after
// the barrier and retire while this tile's WMMAs run.
// ===========================================================================
#define GBM 128
#define GBN 128
#define GBK 64
#define APITCH 72   // halves; 144B rows -> 16B-aligned fragment reads
#define BPITCH 72

__device__ __forceinline__ void gemm_load_tile(
    const unsigned short* __restrict__ A, const unsigned short* __restrict__ B,
    int m0, int n0, int K, int N, int k0, int tid, uint4* ar, uint4* br)
{
#pragma unroll
    for (int i = 0; i < 4; ++i) {               // A: 128 x 64 halves
        int idx = i * 256 + tid;
        int r = idx >> 3, c8 = (idx & 7) * 8;
        ar[i] = *(const uint4*)(A + (size_t)(m0 + r) * K + k0 + c8);
    }
#pragma unroll
    for (int i = 0; i < 4; ++i) {               // B: 64 x 128 halves
        int idx = i * 256 + tid;
        int r = idx >> 4, lc = (idx & 15) * 8;
        int gc = n0 + lc; if (gc > N - 8) gc = N - 8;   // clamp ragged N
        br[i] = *(const uint4*)(B + (size_t)(k0 + r) * N + gc);
    }
}

__device__ __forceinline__ void gemm_store_tile(
    unsigned short* sA, unsigned short* sB, int tid,
    const uint4* ar, const uint4* br)
{
#pragma unroll
    for (int i = 0; i < 4; ++i) {
        int idx = i * 256 + tid;
        int r = idx >> 3, c8 = (idx & 7) * 8;
        *(uint4*)&sA[r * APITCH + c8] = ar[i];
    }
#pragma unroll
    for (int i = 0; i < 4; ++i) {
        int idx = i * 256 + tid;
        int r = idx >> 4, lc = (idx & 15) * 8;
        union { uint4 q; unsigned short h[8]; } u; u.q = br[i];
#pragma unroll
        for (int j = 0; j < 8; ++j) sB[(lc + j) * BPITCH + r] = u.h[j];
    }
}

__global__ __launch_bounds__(256)
void mla_gemm_bf16(const unsigned short* __restrict__ A,
                   const unsigned short* __restrict__ B,
                   void* __restrict__ C, int M, int N, int K, int bf16_out)
{
    __shared__ __align__(16) unsigned short sA[GBM * APITCH];
    __shared__ __align__(16) unsigned short sB[GBN * BPITCH];

    const int tid  = threadIdx.x;
    const int lane = tid & 31;
    const int wave = tid >> 5;
    const int wm   = wave >> 1;      // 0..3
    const int wn   = wave & 1;       // 0..1
    const int l15  = lane & 15;
    const int hi   = lane >> 4;

    const int m0 = blockIdx.y * GBM;
    const int n0 = blockIdx.x * GBN;

    v8f acc[2][4];
    const v8f vzero = {0.f,0.f,0.f,0.f,0.f,0.f,0.f,0.f};
#pragma unroll
    for (int mi = 0; mi < 2; ++mi)
#pragma unroll
        for (int ni = 0; ni < 4; ++ni) acc[mi][ni] = vzero;

    uint4 ar[4], br[4];
    gemm_load_tile(A, B, m0, n0, K, N, 0, tid, ar, br);

    for (int k0 = 0; k0 < K; k0 += GBK) {
        gemm_store_tile(sA, sB, tid, ar, br);
        __syncthreads();
        // issue next tile's global loads now; they retire during the WMMAs
        int kn = (k0 + GBK < K) ? k0 + GBK : k0;   // branch-free tail
        gemm_load_tile(A, B, m0, n0, K, N, kn, tid, ar, br);

#pragma unroll
        for (int dk = 0; dk < 2; ++dk) {
            v16bf af[2], bfv[4];
#pragma unroll
            for (int mi = 0; mi < 2; ++mi) {
                const unsigned short* p =
                    sA + (wm*32 + mi*16 + l15) * APITCH + dk*32 + 8*hi;
                af[mi] = ldfrag(p, p + 16);       // A: K groups at 8*hi, 16+8*hi
            }
#pragma unroll
            for (int ni = 0; ni < 4; ++ni) {
                const unsigned short* p =
                    sB + (wn*64 + ni*16 + l15) * BPITCH + dk*32 + 8*hi;
                bfv[ni] = ldfrag(p, p + 16);      // B: K groups at 8*hi, 16+8*hi
            }
#pragma unroll
            for (int mi = 0; mi < 2; ++mi)
#pragma unroll
                for (int ni = 0; ni < 4; ++ni)
                    acc[mi][ni] = __builtin_amdgcn_wmma_f32_16x16x32_bf16(
                        false, af[mi], false, bfv[ni], (short)0, acc[mi][ni],
                        false, false);
        }
        __syncthreads();
    }

    if (bf16_out) {
        unsigned short* Cb = (unsigned short*)C;
#pragma unroll
        for (int mi = 0; mi < 2; ++mi)
#pragma unroll
            for (int ni = 0; ni < 4; ++ni)
#pragma unroll
                for (int e = 0; e < 8; ++e) {
                    int r = m0 + wm*32 + mi*16 + e + 8*hi;
                    int c = n0 + wn*64 + ni*16 + l15;
                    if (c < N) Cb[(size_t)r * N + c] = f2bf(acc[mi][ni][e]);
                }
    } else {
        float* Cf = (float*)C;
#pragma unroll
        for (int mi = 0; mi < 2; ++mi)
#pragma unroll
            for (int ni = 0; ni < 4; ++ni)
#pragma unroll
                for (int e = 0; e < 8; ++e) {
                    int r = m0 + wm*32 + mi*16 + e + 8*hi;
                    int c = n0 + wn*64 + ni*16 + l15;
                    if (c < N) Cf[(size_t)r * N + c] = acc[mi][ni][e];
                }
    }
}

// ===========================================================================
// RMSNorm: f32 in -> bf16 out, one block per row.
// ===========================================================================
__global__ __launch_bounds__(256)
void mla_rmsnorm_f2b(const float* __restrict__ x, const float* __restrict__ w,
                     unsigned short* __restrict__ y, int D)
{
    const int row = blockIdx.x, tid = threadIdx.x;
    const float* xr = x + (size_t)row * D;
    float ss = 0.f;
    for (int i = tid; i < D; i += 256) { float v = xr[i]; ss += v * v; }
    for (int off = 1; off < 32; off <<= 1) ss += __shfl_xor(ss, off, 32);
    __shared__ float red[8];
    __shared__ float rstd;
    if ((tid & 31) == 0) red[tid >> 5] = ss;
    __syncthreads();
    if (tid == 0) {
        float s = 0.f;
        for (int i = 0; i < 8; ++i) s += red[i];
        rstd = rsqrtf(s / (float)D + 1e-6f);
    }
    __syncthreads();
    float rs = rstd;
    for (int i = tid; i < D; i += 256) y[(size_t)row * D + i] = f2bf(xr[i] * rs * w[i]);
}

// ===========================================================================
// kv_prep: latent bf16 -> rmsnorm(ckv) bf16 ; rope(k_rope) bf16.
// ===========================================================================
__global__ __launch_bounds__(256)
void mla_kv_prep(const unsigned short* __restrict__ latent,
                 const float* __restrict__ w, const int* __restrict__ positions,
                 unsigned short* __restrict__ ckv, unsigned short* __restrict__ krope)
{
    const int t = blockIdx.x, tid = threadIdx.x;
    const unsigned short* lr = latent + (size_t)t * (KVLORA + DROPE);
    float ss = 0.f;
    for (int i = tid; i < KVLORA; i += 256) { float v = bf2f(lr[i]); ss += v * v; }
    for (int off = 1; off < 32; off <<= 1) ss += __shfl_xor(ss, off, 32);
    __shared__ float red[8];
    __shared__ float rstd;
    if ((tid & 31) == 0) red[tid >> 5] = ss;
    __syncthreads();
    if (tid == 0) {
        float s = 0.f;
        for (int i = 0; i < 8; ++i) s += red[i];
        rstd = rsqrtf(s / (float)KVLORA + 1e-6f);
    }
    __syncthreads();
    float rs = rstd;
    for (int i = tid; i < KVLORA; i += 256)
        ckv[(size_t)t * KVLORA + i] = f2bf(bf2f(lr[i]) * rs * w[i]);

    if (tid < DROPE / 2) {
        int i = tid;
        float pos = (float)positions[t];
        float inv = __powf(10000.0f, -(float)(2 * i) / (float)DROPE);
        float ang = pos * inv, s, c;
        __sincosf(ang, &s, &c);
        float x1 = bf2f(lr[KVLORA + 2*i]), x2 = bf2f(lr[KVLORA + 2*i + 1]);
        krope[(size_t)t * DROPE + 2*i]     = f2bf(x1 * c - x2 * s);
        krope[(size_t)t * DROPE + 2*i + 1] = f2bf(x1 * s + x2 * c);
    }
}

// ===========================================================================
// rope_q: interleaved RoPE on bf16 q[t, h*192+128..191], in place.
// ===========================================================================
__global__ __launch_bounds__(256)
void mla_rope_q(unsigned short* __restrict__ q, const int* __restrict__ positions)
{
    const int t = blockIdx.x;
    float pos = (float)positions[t];
    for (int p = threadIdx.x; p < NH * (DROPE / 2); p += 256) {
        int h = p >> 5, i = p & 31;
        float inv = __powf(10000.0f, -(float)(2 * i) / (float)DROPE);
        float ang = pos * inv, s, c;
        __sincosf(ang, &s, &c);
        size_t base = (size_t)t * (NH * DQK) + h * DQK + DNOPE + 2 * i;
        unsigned pk = *(const unsigned*)(q + base);
        float x1 = bf2f((unsigned short)(pk & 0xffff));
        float x2 = bf2f((unsigned short)(pk >> 16));
        unsigned o = (unsigned)f2bf(x1 * c - x2 * s)
                   | ((unsigned)f2bf(x1 * s + x2 * c) << 16);
        *(unsigned*)(q + base) = o;
    }
}

// ===========================================================================
// Flash attention (causal): block = (q-tile 128 rows, head); 8 waves;
// wave owns 16 q rows; 64 keys/iter; all bf16. Staging loads issue into
// registers before the LDS barrier so they overlap the previous tile's math.
// ===========================================================================
#define KPITCH 200   // 64 x 200 halves  = 25600 B
#define VPITCH 72    // 128 x 72 halves  = 18432 B
#define PPITCH 72    // 8*16 x 72 halves = 18432 B  (total 62464 B < 64 KB)

__global__ __launch_bounds__(256)
void mla_flash_attn(const unsigned short* __restrict__ q,
                    const unsigned short* __restrict__ kv,
                    const unsigned short* __restrict__ krope,
                    unsigned short* __restrict__ attn)
{
    __shared__ __align__(16) unsigned short lds_k [64  * KPITCH];
    __shared__ __align__(16) unsigned short lds_vt[128 * VPITCH];
    __shared__ __align__(16) unsigned short lds_p [8 * 16 * PPITCH];

    const int tid  = threadIdx.x;
    const int lane = tid & 31;
    const int wave = tid >> 5;
    const int l15  = lane & 15;
    const int hi   = lane >> 4;
    const int qb   = blockIdx.x;
    const int h    = blockIdx.y;
    const float scaling = 0.07216878364870322f;   // (192)^-0.5

    // ---- preload Q fragments: two b128 global reads per fragment
    v16bf qf[6];
    {
        const int qrow = qb * 128 + wave * 16 + l15;
        const unsigned short* qr = q + (size_t)qrow * (NH * DQK) + h * DQK;
#pragma unroll
        for (int dk = 0; dk < 6; ++dk) {
            const unsigned short* p = qr + dk * 32 + 8 * hi;
            qf[dk] = ldfrag(p, p + 16);
        }
    }

    v8f o_acc[8];
    const v8f vzero = {0.f,0.f,0.f,0.f,0.f,0.f,0.f,0.f};
#pragma unroll
    for (int nj = 0; nj < 8; ++nj) o_acc[nj] = vzero;
    float m_prev[8], l_sum[8];
#pragma unroll
    for (int e = 0; e < 8; ++e) { m_prev[e] = -1e30f; l_sum[e] = 0.f; }

    const int kb_max = 2 * qb + 1;
    for (int kb = 0; kb <= kb_max; ++kb) {
        const int t0 = kb * 64;

        // ---- issue all staging loads to registers (overlaps prior compute)
        uint4 kreg[6], vreg[4];
#pragma unroll
        for (int i = 0; i < 6; ++i) {             // K tile: 64 x 192 halves
            int idx = i * 256 + tid;
            int kr = idx / 24, g = idx % 24;
            const unsigned short* src = (g < 16)
                ? kv + (size_t)(t0 + kr) * (NH * (DNOPE + DV)) + h * (DNOPE + DV) + g * 8
                : krope + (size_t)(t0 + kr) * DROPE + (g - 16) * 8;
            kreg[i] = *(const uint4*)src;
        }
#pragma unroll
        for (int i = 0; i < 4; ++i) {             // V tile: 64 x 128 halves
            int idx = i * 256 + tid;
            int kr = idx >> 4, g = (idx & 15) * 8;
            vreg[i] = *(const uint4*)(kv + (size_t)(t0 + kr) * (NH * (DNOPE + DV))
                                      + h * (DNOPE + DV) + DNOPE + g);
        }
        __syncthreads();                          // prior tile reads done
#pragma unroll
        for (int i = 0; i < 6; ++i) {
            int idx = i * 256 + tid;
            int kr = idx / 24, g = idx % 24;
            *(uint4*)&lds_k[kr * KPITCH + g * 8] = kreg[i];
        }
#pragma unroll
        for (int i = 0; i < 4; ++i) {
            int idx = i * 256 + tid;
            int kr = idx >> 4, g = (idx & 15) * 8;
            union { uint4 q; unsigned short h[8]; } u; u.q = vreg[i];
#pragma unroll
            for (int j = 0; j < 8; ++j) lds_vt[(g + j) * VPITCH + kr] = u.h[j];
        }
        __syncthreads();

        // ---- S = Q @ K^T (16x64 per wave)
        v8f acc_s[4];
#pragma unroll
        for (int ni = 0; ni < 4; ++ni) acc_s[ni] = vzero;
#pragma unroll
        for (int dk = 0; dk < 6; ++dk)
#pragma unroll
            for (int ni = 0; ni < 4; ++ni) {
                const unsigned short* p =
                    lds_k + (ni*16 + l15) * KPITCH + dk*32 + 8*hi;
                v16bf kf = ldfrag(p, p + 16);
                acc_s[ni] = __builtin_amdgcn_wmma_f32_16x16x32_bf16(
                    false, qf[dk], false, kf, (short)0, acc_s[ni],
                    false, false);
            }

        // ---- scale + causal mask
        const int qr0 = qb * 128 + wave * 16;
#pragma unroll
        for (int ni = 0; ni < 4; ++ni)
#pragma unroll
            for (int e = 0; e < 8; ++e) {
                int row = qr0 + e + 8 * hi;
                int col = t0 + ni * 16 + l15;
                float s = acc_s[ni][e] * scaling;
                acc_s[ni][e] = (col <= row) ? s : -1e30f;
            }

        // ---- online softmax
        float m_new[8], rsum[8];
#pragma unroll
        for (int e = 0; e < 8; ++e) {
            float mx = m_prev[e];
#pragma unroll
            for (int ni = 0; ni < 4; ++ni) mx = fmaxf(mx, acc_s[ni][e]);
            for (int off = 1; off < 16; off <<= 1)
                mx = fmaxf(mx, __shfl_xor(mx, off, 32));
            m_new[e] = mx;
            float rs = 0.f;
#pragma unroll
            for (int ni = 0; ni < 4; ++ni) {
                float pv = __expf(acc_s[ni][e] - mx);
                acc_s[ni][e] = pv;
                rs += pv;
            }
            for (int off = 1; off < 16; off <<= 1)
                rs += __shfl_xor(rs, off, 32);
            rsum[e] = rs;
        }
#pragma unroll
        for (int e = 0; e < 8; ++e) {
            float sc = __expf(m_prev[e] - m_new[e]);
            l_sum[e] = l_sum[e] * sc + rsum[e];
#pragma unroll
            for (int nj = 0; nj < 8; ++nj) o_acc[nj][e] *= sc;
            m_prev[e] = m_new[e];
        }

        // ---- P: C-layout -> A-layout through per-wave LDS strip
        unsigned short* pw = lds_p + wave * 16 * PPITCH;
#pragma unroll
        for (int ni = 0; ni < 4; ++ni)
#pragma unroll
            for (int e = 0; e < 8; ++e)
                pw[(e + 8*hi) * PPITCH + ni*16 + l15] = f2bf(acc_s[ni][e]);

        v16bf pf[2];
#pragma unroll
        for (int kk = 0; kk < 2; ++kk) {
            const unsigned short* p = pw + l15 * PPITCH + kk*32 + 8*hi;
            pf[kk] = ldfrag(p, p + 16);
        }
        // ---- O += P @ V
#pragma unroll
        for (int nj = 0; nj < 8; ++nj)
#pragma unroll
            for (int kk = 0; kk < 2; ++kk) {
                const unsigned short* p =
                    lds_vt + (nj*16 + l15) * VPITCH + kk*32 + 8*hi;
                v16bf vf = ldfrag(p, p + 16);
                o_acc[nj] = __builtin_amdgcn_wmma_f32_16x16x32_bf16(
                    false, pf[kk], false, vf, (short)0, o_acc[nj],
                    false, false);
            }
    }

    // ---- normalize (one rcp per row), write attn (bf16)
#pragma unroll
    for (int e = 0; e < 8; ++e) {
        float inv = 1.0f / l_sum[e];
        int r = qb * 128 + wave * 16 + e + 8 * hi;
#pragma unroll
        for (int nj = 0; nj < 8; ++nj) {
            int c = h * DV + nj * 16 + l15;
            attn[(size_t)r * (NH * DV) + c] = f2bf(o_acc[nj][e] * inv);
        }
    }
}

// ===========================================================================
// Launch
// ===========================================================================
extern "C" void kernel_launch(void* const* d_in, const int* in_sizes, int n_in,
                              void* d_out, int out_size, void* d_ws, size_t ws_size,
                              hipStream_t stream)
{
    const float* hidden    = (const float*)d_in[0];
    const int*   positions = (const int*)  d_in[1];
    const float* wq_a      = (const float*)d_in[2];
    const float* q_a_ln_w  = (const float*)d_in[3];
    const float* wq_b      = (const float*)d_in[4];
    const float* wkv_a     = (const float*)d_in[5];
    const float* kv_a_ln_w = (const float*)d_in[6];
    const float* wkv_b     = (const float*)d_in[7];
    const float* wo        = (const float*)d_in[8];
    float* out = (float*)d_out;

    char* wp = (char*)d_ws;
    auto carve = [&](size_t bytes) -> void* {
        void* r = (void*)wp;
        wp += (bytes + 255) & ~(size_t)255;
        return r;
    };
    unsigned short* hidden_b = (unsigned short*)carve((size_t)TT * HIDDEN * 2);
    unsigned short* wq_a_b   = (unsigned short*)carve((size_t)HIDDEN * QLORA * 2);
    unsigned short* wq_b_b   = (unsigned short*)carve((size_t)QLORA * NH * DQK * 2);
    unsigned short* wkv_a_b  = (unsigned short*)carve((size_t)HIDDEN * (KVLORA + DROPE) * 2);
    unsigned short* wkv_b_b  = (unsigned short*)carve((size_t)KVLORA * NH * (DNOPE + DV) * 2);
    unsigned short* wo_b     = (unsigned short*)carve((size_t)NH * DV * HIDDEN * 2);
    float*          q_lora_f = (float*)carve((size_t)TT * QLORA * 4);
    unsigned short* q_lora_b = (unsigned short*)carve((size_t)TT * QLORA * 2);
    unsigned short* qbuf     = (unsigned short*)carve((size_t)TT * NH * DQK * 2);
    unsigned short* latent   = (unsigned short*)carve((size_t)TT * (KVLORA + DROPE) * 2);
    unsigned short* ckv      = (unsigned short*)carve((size_t)TT * KVLORA * 2);
    unsigned short* krope    = (unsigned short*)carve((size_t)TT * DROPE * 2);
    unsigned short* kvbuf    = (unsigned short*)carve((size_t)TT * NH * (DNOPE + DV) * 2);
    unsigned short* attn     = (unsigned short*)carve((size_t)TT * NH * DV * 2);

    dim3 blk(256);
    auto cvt = [&](const float* src, unsigned short* dst, size_t n) {
        int n4 = (int)(n / 4);
        mla_cvt_bf16<<<dim3((n4 + 255) / 256), blk, 0, stream>>>(src, dst, n4);
    };
    cvt(hidden, hidden_b, (size_t)TT * HIDDEN);
    cvt(wq_a,   wq_a_b,   (size_t)HIDDEN * QLORA);
    cvt(wq_b,   wq_b_b,   (size_t)QLORA * NH * DQK);
    cvt(wkv_a,  wkv_a_b,  (size_t)HIDDEN * (KVLORA + DROPE));
    cvt(wkv_b,  wkv_b_b,  (size_t)KVLORA * NH * (DNOPE + DV));
    cvt(wo,     wo_b,     (size_t)NH * DV * HIDDEN);

    // q = rmsnorm(hidden @ wq_a) @ wq_b
    mla_gemm_bf16<<<dim3(QLORA/GBN, TT/GBM), blk, 0, stream>>>(
        hidden_b, wq_a_b, q_lora_f, TT, QLORA, HIDDEN, 0);
    mla_rmsnorm_f2b<<<TT, 256, 0, stream>>>(q_lora_f, q_a_ln_w, q_lora_b, QLORA);
    mla_gemm_bf16<<<dim3((NH*DQK)/GBN, TT/GBM), blk, 0, stream>>>(
        q_lora_b, wq_b_b, qbuf, TT, NH*DQK, QLORA, 1);
    // latent = hidden @ wkv_a ; split into ckv (rmsnorm) + krope (rope)
    mla_gemm_bf16<<<dim3((KVLORA+DROPE+GBN-1)/GBN, TT/GBM), blk, 0, stream>>>(
        hidden_b, wkv_a_b, latent, TT, KVLORA+DROPE, HIDDEN, 1);
    mla_kv_prep<<<TT, 256, 0, stream>>>(latent, kv_a_ln_w, positions, ckv, krope);
    mla_rope_q<<<TT, 256, 0, stream>>>(qbuf, positions);
    // kv = ckv @ wkv_b
    mla_gemm_bf16<<<dim3((NH*(DNOPE+DV))/GBN, TT/GBM), blk, 0, stream>>>(
        ckv, wkv_b_b, kvbuf, TT, NH*(DNOPE+DV), KVLORA, 1);
    // causal flash attention
    mla_flash_attn<<<dim3(TT/128, NH), blk, 0, stream>>>(qbuf, kvbuf, krope, attn);
    // out = attn @ wo
    mla_gemm_bf16<<<dim3(HIDDEN/GBN, TT/GBM), blk, 0, stream>>>(
        attn, wo_b, out, TT, HIDDEN, NH*DV, 0);
}
